// ActionTransformer3_82033875353846
// MI455X (gfx1250) — compile-verified, matrix operand order checked by reading, not run on hardware
//
#include <hip/hip_runtime.h>

// ---------------------------------------------------------------------------
// Types for CDNA5 WMMA
// ---------------------------------------------------------------------------
typedef __attribute__((ext_vector_type(16))) _Float16 v16h;
typedef __attribute__((ext_vector_type(8)))  _Float16 v8h;
typedef __attribute__((ext_vector_type(8)))  float    v8f;

static __device__ inline v8f wmma_f32_f16(v16h a, v16h b, v8f c) {
  // D = A(16x32,f16) * B(32x16,f16) + C(16x16,f32)
  return __builtin_amdgcn_wmma_f32_16x16x32_f16(
      /*neg_a=*/false, a, /*neg_b=*/false, b,
      /*c_mod=*/(short)0, c, /*reuse_a=*/false, /*reuse_b=*/false);
}

static __device__ inline v8f zero8f() {
  v8f z;
#pragma unroll
  for (int i = 0; i < 8; ++i) z[i] = 0.0f;
  return z;
}
static __device__ inline v16h zero16h() {
  v16h z;
#pragma unroll
  for (int i = 0; i < 16; ++i) z[i] = (_Float16)0.0f;
  return z;
}

// Load one lane's fragment slice: elements {p[0..7], p[16..23]}.
// Caller adds khalf = (lane>=16 ? 8 : 0) so the full wave covers K 0..31
// per the CDNA5 16-bit A-matrix 16x32 VGPR layout. Same pattern serves B
// fragments because B operands are stored transposed (N x K row-major).
static __device__ inline v16h load_frag16(const _Float16* p) {
  v8h lo = *reinterpret_cast<const v8h*>(p);
  v8h hi = *reinterpret_cast<const v8h*>(p + 16);
  return __builtin_shufflevector(lo, hi, 0, 1, 2, 3, 4, 5, 6, 7,
                                 8, 9, 10, 11, 12, 13, 14, 15);
}

// Reductions within 16-lane halves of a wave32 (rows of a WMMA C tile).
static __device__ inline float hmax16(float v) {
#pragma unroll
  for (int m = 1; m < 16; m <<= 1) v = fmaxf(v, __shfl_xor(v, m, 32));
  return v;
}
static __device__ inline float hsum16(float v) {
#pragma unroll
  for (int m = 1; m < 16; m <<= 1) v += __shfl_xor(v, m, 32);
  return v;
}

// ---------------------------------------------------------------------------
// Model dimensions
// ---------------------------------------------------------------------------
#define BATCH 5
#define SEQ   1024
#define TOK   1025   // SEQ + cls
#define TPAD  1056   // token stride for transposed V (16B aligned, > 1055)
#define DM    512
#define DFF   2048
#define NTOKF 51
#define NHEAD 8
#define HD    64
#define NLAYER 4
#define MROWS (BATCH * TOK)     // 5125
#define GALPHA 0.2f

// ---------------------------------------------------------------------------
// Weight conversion: W (K x N, f32) -> Wt (N x K, f16)
// ---------------------------------------------------------------------------
__global__ void k_transpose_f16(const float* __restrict__ W,
                                _Float16* __restrict__ Wt, int K, int N) {
  int idx = blockIdx.x * 256 + threadIdx.x;
  if (idx >= K * N) return;
  int k = idx / N, n = idx % N;
  Wt[(size_t)n * K + k] = (_Float16)W[idx];
}

// ---------------------------------------------------------------------------
// V transpose for attention: V16 [b][t][h*64+d] -> VT [b*8+h][d][t] (TPAD)
// Pad region zero-filled so b128 fragment loads past t=1024 are benign.
// ---------------------------------------------------------------------------
__global__ void k_vtrans(const _Float16* __restrict__ V16,
                         _Float16* __restrict__ VT) {
  int idx = blockIdx.x * 256 + threadIdx.x;
  const int total = BATCH * NHEAD * HD * TPAD;
  if (idx >= total) return;
  int t = idx % TPAD;
  int rem = idx / TPAD;
  int d = rem % HD;
  int bh = rem / HD;
  int h = bh % NHEAD, b = bh / NHEAD;
  _Float16 v = (_Float16)0.0f;
  if (t < TOK) v = V16[((size_t)b * TOK + t) * DM + h * HD + d];
  VT[idx] = v;
}

// ---------------------------------------------------------------------------
// Encoder: h[b,t,:] = (t==0 ? cls : x[b,t-1,:] @ W_enc + b_enc) + pos_emb[t]
// ---------------------------------------------------------------------------
__global__ void k_encode(const float* __restrict__ x,
                         const float* __restrict__ W_enc,
                         const float* __restrict__ b_enc,
                         const float* __restrict__ cls_tok,
                         const float* __restrict__ pos_emb,
                         float* __restrict__ H32, _Float16* __restrict__ H16) {
  int idx = blockIdx.x * 256 + threadIdx.x;
  if (idx >= MROWS * DM) return;
  int n = idx & (DM - 1);
  int bt = idx >> 9;
  int t = bt % TOK, b = bt / TOK;
  float v;
  if (t == 0) {
    v = cls_tok[n] + pos_emb[n];
  } else {
    const float* xr = x + ((size_t)b * SEQ + (t - 1)) * NTOKF;
    float s = b_enc[n];
#pragma unroll 3
    for (int k = 0; k < NTOKF; ++k) s += xr[k] * W_enc[k * DM + n];
    v = s + pos_emb[(size_t)t * DM + n];
  }
  H32[idx] = v;
  H16[idx] = (_Float16)v;
}

// ---------------------------------------------------------------------------
// WMMA GEMM: C[M,N] = A[M,K](f16,row) * Bt[N,K](f16,row) + bias[N]
// MODE 0: f32 out; MODE 1: f16 out; MODE 2: f16 out + ReLU;
// MODE 3: f16 out scaled by 0.125 (Q projection, folds 1/sqrt(HD)).
// Block: 256 threads (8 waves). Block tile 128(M) x 128(N); wave 32x64:
// 2 A-frags x 4 B-frags -> 8 WMMA per K-step (1.5 b128 loads per WMMA).
// ---------------------------------------------------------------------------
template <int MODE>
__global__ __launch_bounds__(256) void k_gemm_wmma(
    const _Float16* __restrict__ A, const _Float16* __restrict__ Bt,
    const float* __restrict__ bias, float* __restrict__ C32,
    _Float16* __restrict__ C16, int M, int N, int K) {
  const int lane = threadIdx.x & 31;
  const int w = threadIdx.x >> 5;
  const int wm = w & 3;   // 4 row bands of 32
  const int wn = w >> 2;  // 2 col groups of 64
  const int mBase = blockIdx.y * 128 + wm * 32;
  const int nBase = blockIdx.x * 128 + wn * 64;
  const int r0 = mBase + (lane & 15);
  const int r1 = r0 + 16;
  const bool ok0 = r0 < M, ok1 = r1 < M;
  const int khalf = (lane >> 4) ? 8 : 0;
  const int nc = nBase + (lane & 15);

  v8f acc[2][4];
#pragma unroll
  for (int i = 0; i < 2; ++i)
#pragma unroll
    for (int j = 0; j < 4; ++j) acc[i][j] = zero8f();

  const _Float16* a0p = A + (size_t)(ok0 ? r0 : 0) * K + khalf;
  const _Float16* a1p = A + (size_t)(ok1 ? r1 : 0) * K + khalf;
  const _Float16* b0p = Bt + (size_t)(nc +  0) * K + khalf;
  const _Float16* b1p = Bt + (size_t)(nc + 16) * K + khalf;
  const _Float16* b2p = Bt + (size_t)(nc + 32) * K + khalf;
  const _Float16* b3p = Bt + (size_t)(nc + 48) * K + khalf;

  for (int k0 = 0; k0 < K; k0 += 32) {
    v16h a0 = ok0 ? load_frag16(a0p + k0) : zero16h();
    v16h a1 = ok1 ? load_frag16(a1p + k0) : zero16h();
    v16h b0 = load_frag16(b0p + k0);
    v16h b1 = load_frag16(b1p + k0);
    v16h b2 = load_frag16(b2p + k0);
    v16h b3 = load_frag16(b3p + k0);
    acc[0][0] = wmma_f32_f16(a0, b0, acc[0][0]);
    acc[0][1] = wmma_f32_f16(a0, b1, acc[0][1]);
    acc[0][2] = wmma_f32_f16(a0, b2, acc[0][2]);
    acc[0][3] = wmma_f32_f16(a0, b3, acc[0][3]);
    acc[1][0] = wmma_f32_f16(a1, b0, acc[1][0]);
    acc[1][1] = wmma_f32_f16(a1, b1, acc[1][1]);
    acc[1][2] = wmma_f32_f16(a1, b2, acc[1][2]);
    acc[1][3] = wmma_f32_f16(a1, b3, acc[1][3]);
  }

#pragma unroll
  for (int j = 0; j < 4; ++j) {
    const int n = nc + 16 * j;
    const float bb = bias[n];
#pragma unroll
    for (int i = 0; i < 2; ++i) {
      const int mO = mBase + 16 * i + 8 * (lane >> 4);
#pragma unroll
      for (int r = 0; r < 8; ++r) {
        int m = mO + r;
        if (m < M) {
          float v = acc[i][j][r] + bb;
          if (MODE == 2) v = fmaxf(v, 0.0f);
          if (MODE == 3) v *= 0.125f;
          if (MODE == 0) C32[(size_t)m * N + n] = v;
          else           C16[(size_t)m * N + n] = (_Float16)v;
        }
      }
    }
  }
}

// ---------------------------------------------------------------------------
// Flash-style attention. One wave per (b, head, 16-query tile).
// Q (pre-scaled by 0.125), K: [b][t][head*64+d]; V transposed: [bh][d][t].
// Online softmax: cross-lane max per step; sum reduction deferred to the end
// (per-lane partials rescaled by the uniform correction factor).
// ---------------------------------------------------------------------------
__global__ __launch_bounds__(32) void k_attention(
    const _Float16* __restrict__ Q, const _Float16* __restrict__ Km,
    const _Float16* __restrict__ VT, _Float16* __restrict__ AO) {
  __shared__ _Float16 pl[16 * 32];

  const int lane = threadIdx.x;
  const int qt = blockIdx.x, head = blockIdx.y, b = blockIdx.z;
  const size_t base = (size_t)b * TOK * DM + head * HD;
  const _Float16* vtb = VT + (size_t)(b * NHEAD + head) * HD * TPAD;
  const int n = lane & 15;
  const int khalf = (lane >> 4) ? 8 : 0;

  // Q fragments for head-dim 0..31 and 32..63
  const int qrow = qt * 16 + n;
  v16h aq0, aq1;
  if (qrow < TOK) {
    const _Float16* qp = Q + base + (size_t)qrow * DM;
    aq0 = load_frag16(qp + khalf);
    aq1 = load_frag16(qp + 32 + khalf);
  } else {
    aq0 = zero16h();
    aq1 = zero16h();
  }

  float rm[8], sacc[8];
  v8f o[4];
#pragma unroll
  for (int r = 0; r < 8; ++r) { rm[r] = -3.0e38f; sacc[r] = 0.0f; }
#pragma unroll
  for (int g = 0; g < 4; ++g) o[g] = zero8f();

  const int nkt = (TOK + 31) / 32;  // 33

  for (int kb = 0; kb < nkt; ++kb) {
    const int tb = kb * 32;
    const int t0 = tb + n, t1 = tb + 16 + n;

    v16h bk0a, bk0b, bk1a, bk1b;
    if (t0 < TOK) {
      const _Float16* kp = Km + base + (size_t)t0 * DM;
      bk0a = load_frag16(kp + khalf);
      bk0b = load_frag16(kp + 32 + khalf);
    } else { bk0a = zero16h(); bk0b = zero16h(); }
    if (t1 < TOK) {
      const _Float16* kp = Km + base + (size_t)t1 * DM;
      bk1a = load_frag16(kp + khalf);
      bk1b = load_frag16(kp + 32 + khalf);
    } else { bk1a = zero16h(); bk1b = zero16h(); }

    v8f s0 = wmma_f32_f16(aq0, bk0a, zero8f());
    s0 = wmma_f32_f16(aq1, bk0b, s0);
    v8f s1 = wmma_f32_f16(aq0, bk1a, zero8f());
    s1 = wmma_f32_f16(aq1, bk1b, s1);

    const bool ok0 = t0 < TOK, ok1 = t1 < TOK;
    const int mrow = 8 * (lane >> 4);
#pragma unroll
    for (int r = 0; r < 8; ++r) {
      float x0 = ok0 ? s0[r] : -1.0e30f;
      float x1 = ok1 ? s1[r] : -1.0e30f;
      float mx = hmax16(fmaxf(x0, x1));       // cross-half-lane row max
      float mnew = fmaxf(rm[r], mx);
      float corr = __expf(rm[r] - mnew);      // uniform across the half
      float p0 = __expf(x0 - mnew);
      float p1 = __expf(x1 - mnew);
      sacc[r] = sacc[r] * corr + (p0 + p1);   // per-lane partial sum
      rm[r] = mnew;
      o[0][r] *= corr; o[1][r] *= corr; o[2][r] *= corr; o[3][r] *= corr;
      int m = mrow + r;
      pl[m * 32 + n] = (_Float16)p0;
      pl[m * 32 + 16 + n] = (_Float16)p1;
    }
    __syncthreads();

    // P (16x32) as A-fragment from LDS
    v16h ap = load_frag16(&pl[(lane & 15) * 32 + khalf]);

    // V fragments from transposed V: contiguous b128 loads
#pragma unroll
    for (int g = 0; g < 4; ++g) {
      const int d = g * 16 + n;
      v16h bv = load_frag16(vtb + (size_t)d * TPAD + tb + khalf);
      o[g] = wmma_f32_f16(ap, bv, o[g]);
    }
    __syncthreads();
  }

  // Final: reduce deferred row sums once, normalize, store f16
  const int mrow = 8 * (lane >> 4);
#pragma unroll
  for (int r = 0; r < 8; ++r) {
    float rl = hsum16(sacc[r]);
    int t = qt * 16 + mrow + r;
    if (t < TOK) {
      float inv = 1.0f / rl;
#pragma unroll
      for (int g = 0; g < 4; ++g) {
        AO[base + (size_t)t * DM + g * 16 + n] = (_Float16)(o[g][r] * inv);
      }
    }
  }
}

// ---------------------------------------------------------------------------
// Residual add + LayerNorm, writes f32 and f16 copies. One row per block.
// ---------------------------------------------------------------------------
__global__ __launch_bounds__(128) void k_resid_ln(
    const float* X, const float* R, const float* __restrict__ gam,
    const float* __restrict__ bet, float* H32, _Float16* __restrict__ H16) {
  __shared__ float red[8];
  const int row = blockIdx.x;
  const int tid = threadIdx.x;
  const size_t basep = (size_t)row * DM;

  float y[4];
  float s = 0.0f, sq = 0.0f;
#pragma unroll
  for (int j = 0; j < 4; ++j) {
    int idx = tid * 4 + j;
    float v = X[basep + idx] + R[basep + idx];
    y[j] = v;
    s += v;
    sq += v * v;
  }
#pragma unroll
  for (int m = 1; m < 32; m <<= 1) {
    s += __shfl_xor(s, m, 32);
    sq += __shfl_xor(sq, m, 32);
  }
  int wv = tid >> 5;
  if ((tid & 31) == 0) { red[wv] = s; red[4 + wv] = sq; }
  __syncthreads();
  s = red[0] + red[1] + red[2] + red[3];
  sq = red[4] + red[5] + red[6] + red[7];
  float mean = s * (1.0f / DM);
  float var = sq * (1.0f / DM) - mean * mean;
  float rstd = rsqrtf(var + 1e-5f);
#pragma unroll
  for (int j = 0; j < 4; ++j) {
    int idx = tid * 4 + j;
    float o = (y[j] - mean) * rstd * gam[idx] + bet[idx];
    H32[basep + idx] = o;
    H16[basep + idx] = (_Float16)o;
  }
}

// ---------------------------------------------------------------------------
// Classifier head on cls tokens: (cls @ Wd1 + bd1) @ Wd2 + bd2 -> out (5,7)
// ---------------------------------------------------------------------------
__global__ __launch_bounds__(256) void k_head(
    const float* __restrict__ H32, const float* __restrict__ Wd1,
    const float* __restrict__ bd1, const float* __restrict__ Wd2,
    const float* __restrict__ bd2, float* __restrict__ out) {
  __shared__ float t1[BATCH][256];
  const int tid = threadIdx.x;
  for (int idx = tid; idx < BATCH * 256; idx += 256) {
    int b = idx / 256, j = idx % 256;
    const float* cls = H32 + (size_t)b * TOK * DM;  // t = 0
    float s = bd1[j];
    for (int k = 0; k < DM; ++k) s += cls[k] * Wd1[k * 256 + j];
    t1[b][j] = s;
  }
  __syncthreads();
  for (int idx = tid; idx < BATCH * 7; idx += 256) {
    int b = idx / 7, c = idx % 7;
    float s = bd2[c];
    for (int j = 0; j < 256; ++j) s += t1[b][j] * Wd2[j * 7 + c];
    out[idx] = s;
  }
}

// ---------------------------------------------------------------------------
// GAT branch: 5120 independent 5-node graphs. 4 graphs per 128-thread block,
// one wave32 per graph, all intermediates in LDS.
// ---------------------------------------------------------------------------
#define GPB 4
__global__ __launch_bounds__(128) void k_gat(
    const float* __restrict__ x, const int* __restrict__ adj,
    const float* __restrict__ Wg, const float* __restrict__ ag,
    const float* __restrict__ Wgo, const float* __restrict__ ago,
    float* __restrict__ gout) {
  __shared__ float hbuf[GPB][5][192];
  __shared__ float hcat[GPB][5][192];
  __shared__ float attb[GPB][3][25];
  __shared__ float scb[GPB][32];
  __shared__ float h2b[GPB][25];
  __shared__ float att2b[GPB][25];
  __shared__ float sc2b[GPB][12];
  __shared__ float outb[GPB][25];
  __shared__ float lseb[GPB][5];
  __shared__ int maskb[25];

  const int g = threadIdx.x >> 5;
  const int lane = threadIdx.x & 31;
  const int gid = blockIdx.x * GPB + g;
  const int s = gid / 5, c = gid % 5;

  if (threadIdx.x < 25) maskb[threadIdx.x] = adj[threadIdx.x] > 0 ? 1 : 0;

  // Node features: feat[i][j] = x[i][s][c*3+j] (uniform across the wave)
  float feat[5][3];
#pragma unroll
  for (int i = 0; i < 5; ++i)
#pragma unroll
    for (int j = 0; j < 3; ++j)
      feat[i][j] = x[((size_t)i * SEQ + s) * NTOKF + c * 3 + j];
  __syncthreads();

  // Hidden: h[i][u] for u = head*64 + f, 3 heads
  for (int u = lane; u < 192; u += 32) {
    int hh = u >> 6, f = u & 63;
    const float* w = Wg + hh * 192 + f;  // Wg[hh][j][f], stride 64 in j
#pragma unroll
    for (int i = 0; i < 5; ++i) {
      hbuf[g][i][u] =
          feat[i][0] * w[0] + feat[i][1] * w[64] + feat[i][2] * w[128];
    }
  }
  __syncthreads();

  // Attention scores (src/dst) per head/node: 30 values
  for (int idx = lane; idx < 30; idx += 32) {
    int hh = idx / 10, rem = idx % 10;
    int which = rem / 5, node = rem % 5;
    const float* av = ag + hh * 128 + which * 64;
    float sum = 0.0f;
    for (int f = 0; f < 64; ++f) sum += hbuf[g][node][hh * 64 + f] * av[f];
    scb[g][idx] = sum;
  }
  __syncthreads();

  // Masked softmax rows per (head, i): 15 rows
  for (int idx = lane; idx < 15; idx += 32) {
    int hh = idx / 5, i = idx % 5;
    float src = scb[g][hh * 10 + i];
    float e[5], mx = -3.0e38f;
#pragma unroll
    for (int j = 0; j < 5; ++j) {
      float ee = src + scb[g][hh * 10 + 5 + j];
      ee = ee > 0.0f ? ee : GALPHA * ee;  // leaky_relu
      if (!maskb[i * 5 + j]) ee = -9.0e15f;
      e[j] = ee;
      mx = fmaxf(mx, ee);
    }
    float sum = 0.0f;
#pragma unroll
    for (int j = 0; j < 5; ++j) { e[j] = __expf(e[j] - mx); sum += e[j]; }
    float inv = 1.0f / sum;
#pragma unroll
    for (int j = 0; j < 5; ++j) attb[g][hh][i * 5 + j] = e[j] * inv;
  }
  __syncthreads();

  // hcat[i][u] = elu(sum_j att[hh][i][j] * h[j][u])
  for (int u = lane; u < 192; u += 32) {
    int hh = u >> 6;
#pragma unroll
    for (int i = 0; i < 5; ++i) {
      float sum = 0.0f;
#pragma unroll
      for (int j = 0; j < 5; ++j) sum += attb[g][hh][i * 5 + j] * hbuf[g][j][u];
      hcat[g][i][u] = sum > 0.0f ? sum : __expf(sum) - 1.0f;  // elu
    }
  }
  __syncthreads();

  // Output GAT layer: h2 = hcat @ Wgo (192 x 5)
  for (int idx = lane; idx < 25; idx += 32) {
    int i = idx / 5, cl = idx % 5;
    float sum = 0.0f;
    for (int u = 0; u < 192; ++u) sum += hcat[g][i][u] * Wgo[u * 5 + cl];
    h2b[g][idx] = sum;
  }
  __syncthreads();

  for (int idx = lane; idx < 10; idx += 32) {
    int which = idx / 5, node = idx % 5;
    float sum = 0.0f;
#pragma unroll
    for (int cc = 0; cc < 5; ++cc) sum += h2b[g][node * 5 + cc] * ago[which * 5 + cc];
    sc2b[g][idx] = sum;
  }
  __syncthreads();

  for (int idx = lane; idx < 5; idx += 32) {
    int i = idx;
    float src = sc2b[g][i];
    float e[5], mx = -3.0e38f;
#pragma unroll
    for (int j = 0; j < 5; ++j) {
      float ee = src + sc2b[g][5 + j];
      ee = ee > 0.0f ? ee : GALPHA * ee;
      if (!maskb[i * 5 + j]) ee = -9.0e15f;
      e[j] = ee;
      mx = fmaxf(mx, ee);
    }
    float sum = 0.0f;
#pragma unroll
    for (int j = 0; j < 5; ++j) { e[j] = __expf(e[j] - mx); sum += e[j]; }
    float inv = 1.0f / sum;
#pragma unroll
    for (int j = 0; j < 5; ++j) att2b[g][i * 5 + j] = e[j] * inv;
  }
  __syncthreads();

  for (int idx = lane; idx < 25; idx += 32) {
    int i = idx / 5, cl = idx % 5;
    float sum = 0.0f;
#pragma unroll
    for (int j = 0; j < 5; ++j) sum += att2b[g][i * 5 + j] * h2b[g][j * 5 + cl];
    outb[g][idx] = sum > 0.0f ? sum : __expf(sum) - 1.0f;
  }
  __syncthreads();

  for (int idx = lane; idx < 5; idx += 32) {
    float mx = -3.0e38f;
#pragma unroll
    for (int cc = 0; cc < 5; ++cc) mx = fmaxf(mx, outb[g][idx * 5 + cc]);
    float sum = 0.0f;
#pragma unroll
    for (int cc = 0; cc < 5; ++cc) sum += __expf(outb[g][idx * 5 + cc] - mx);
    lseb[g][idx] = mx + __logf(sum);
  }
  __syncthreads();

  for (int idx = lane; idx < 25; idx += 32) {
    gout[(size_t)gid * 25 + idx] = outb[g][idx] - lseb[g][idx / 5];
  }
}

// ---------------------------------------------------------------------------
// Host launch
// ---------------------------------------------------------------------------
extern "C" void kernel_launch(void* const* d_in, const int* in_sizes, int n_in,
                              void* d_out, int out_size, void* d_ws,
                              size_t ws_size, hipStream_t stream) {
  const float* x       = (const float*)d_in[0];
  const int*   adj     = (const int*)d_in[1];
  const float* W_enc   = (const float*)d_in[2];
  const float* b_enc   = (const float*)d_in[3];
  const float* cls_tok = (const float*)d_in[4];
  const float* pos_emb = (const float*)d_in[5];
  const float* Wq = (const float*)d_in[6];
  const float* bq = (const float*)d_in[7];
  const float* Wk = (const float*)d_in[8];
  const float* bk = (const float*)d_in[9];
  const float* Wv = (const float*)d_in[10];
  const float* bv = (const float*)d_in[11];
  const float* Wo = (const float*)d_in[12];
  const float* bo = (const float*)d_in[13];
  const float* W1 = (const float*)d_in[14];
  const float* b1 = (const float*)d_in[15];
  const float* W2 = (const float*)d_in[16];
  const float* b2 = (const float*)d_in[17];
  const float* g1 = (const float*)d_in[18];
  const float* be1 = (const float*)d_in[19];
  const float* g2 = (const float*)d_in[20];
  const float* be2 = (const float*)d_in[21];
  const float* Wd1 = (const float*)d_in[22];
  const float* bd1 = (const float*)d_in[23];
  const float* Wd2 = (const float*)d_in[24];
  const float* bd2 = (const float*)d_in[25];
  const float* Wg = (const float*)d_in[26];
  const float* ag = (const float*)d_in[27];
  const float* Wgo = (const float*)d_in[28];
  const float* ago = (const float*)d_in[29];

  // Workspace layout
  const size_t HSZ = (size_t)MROWS * DM;  // 2,624,000 elements
  const size_t VTSZ = (size_t)BATCH * NHEAD * HD * TPAD;  // 2,703,360
  char* ws = (char*)d_ws;
  size_t off = 0;
  auto alloc = [&](size_t bytes) {
    void* p = ws + off;
    off += (bytes + 255) & ~(size_t)255;
    return p;
  };
  float*    H32   = (float*)alloc(HSZ * 4);
  _Float16* H16   = (_Float16*)alloc(HSZ * 2);
  float*    TMP32 = (float*)alloc(HSZ * 4);
  _Float16* SC16  = (_Float16*)alloc(4 * HSZ * 2);  // Q/K/V/AO, reused for FF
  _Float16* Q16  = SC16;
  _Float16* K16  = SC16 + HSZ;
  _Float16* V16  = SC16 + 2 * HSZ;
  _Float16* AO16 = SC16 + 3 * HSZ;
  _Float16* FF16 = SC16;  // 5*1025*2048 elems == 4*HSZ; Q/K/V/AO dead by then
  _Float16* VT16 = (_Float16*)alloc(VTSZ * 2);
  _Float16* WqT = (_Float16*)alloc((size_t)NLAYER * DM * DM * 2);
  _Float16* WkT = (_Float16*)alloc((size_t)NLAYER * DM * DM * 2);
  _Float16* WvT = (_Float16*)alloc((size_t)NLAYER * DM * DM * 2);
  _Float16* WoT = (_Float16*)alloc((size_t)NLAYER * DM * DM * 2);
  _Float16* W1T = (_Float16*)alloc((size_t)NLAYER * DM * DFF * 2);
  _Float16* W2T = (_Float16*)alloc((size_t)NLAYER * DFF * DM * 2);

  // Convert weights to f16, transposed (N x K)
  const int nSq = DM * DM;   // 262144
  const int nFf = DM * DFF;  // 1048576
  for (int l = 0; l < NLAYER; ++l) {
    k_transpose_f16<<<(nSq + 255) / 256, 256, 0, stream>>>(Wq + (size_t)l * nSq, WqT + (size_t)l * nSq, DM, DM);
    k_transpose_f16<<<(nSq + 255) / 256, 256, 0, stream>>>(Wk + (size_t)l * nSq, WkT + (size_t)l * nSq, DM, DM);
    k_transpose_f16<<<(nSq + 255) / 256, 256, 0, stream>>>(Wv + (size_t)l * nSq, WvT + (size_t)l * nSq, DM, DM);
    k_transpose_f16<<<(nSq + 255) / 256, 256, 0, stream>>>(Wo + (size_t)l * nSq, WoT + (size_t)l * nSq, DM, DM);
    k_transpose_f16<<<(nFf + 255) / 256, 256, 0, stream>>>(W1 + (size_t)l * nFf, W1T + (size_t)l * nFf, DM, DFF);
    k_transpose_f16<<<(nFf + 255) / 256, 256, 0, stream>>>(W2 + (size_t)l * nFf, W2T + (size_t)l * nFf, DFF, DM);
  }

  // Encoder
  {
    int n = MROWS * DM;
    k_encode<<<(n + 255) / 256, 256, 0, stream>>>(x, W_enc, b_enc, cls_tok,
                                                  pos_emb, H32, H16);
  }

  const int MT = (MROWS + 127) / 128;  // 41
  for (int l = 0; l < NLAYER; ++l) {
    // Q (scaled by 1/sqrt(HD)), K, V projections (f16 out with bias)
    k_gemm_wmma<3><<<dim3(DM / 128, MT), 256, 0, stream>>>(
        H16, WqT + (size_t)l * nSq, bq + (size_t)l * DM, nullptr, Q16, MROWS, DM, DM);
    k_gemm_wmma<1><<<dim3(DM / 128, MT), 256, 0, stream>>>(
        H16, WkT + (size_t)l * nSq, bk + (size_t)l * DM, nullptr, K16, MROWS, DM, DM);
    k_gemm_wmma<1><<<dim3(DM / 128, MT), 256, 0, stream>>>(
        H16, WvT + (size_t)l * nSq, bv + (size_t)l * DM, nullptr, V16, MROWS, DM, DM);

    // Transpose V for coalesced b128 B-fragment loads in attention
    k_vtrans<<<((int)VTSZ + 255) / 256, 256, 0, stream>>>(V16, VT16);

    // Attention -> AO16
    k_attention<<<dim3((TOK + 15) / 16, NHEAD, BATCH), 32, 0, stream>>>(
        Q16, K16, VT16, AO16);

    // Output projection (f32 out with bias)
    k_gemm_wmma<0><<<dim3(DM / 128, MT), 256, 0, stream>>>(
        AO16, WoT + (size_t)l * nSq, bo + (size_t)l * DM, TMP32, nullptr, MROWS, DM, DM);
    k_resid_ln<<<MROWS, 128, 0, stream>>>(H32, TMP32, g1 + (size_t)l * DM,
                                          be1 + (size_t)l * DM, H32, H16);

    // FFN
    k_gemm_wmma<2><<<dim3(DFF / 128, MT), 256, 0, stream>>>(
        H16, W1T + (size_t)l * nFf, b1 + (size_t)l * DFF, nullptr, FF16, MROWS, DFF, DM);
    k_gemm_wmma<0><<<dim3(DM / 128, MT), 256, 0, stream>>>(
        FF16, W2T + (size_t)l * nFf, b2 + (size_t)l * DM, TMP32, nullptr, MROWS, DM, DFF);
    k_resid_ln<<<MROWS, 128, 0, stream>>>(H32, TMP32, g2 + (size_t)l * DM,
                                          be2 + (size_t)l * DM, H32, H16);
  }

  // Classifier head -> d_out[0..35)
  k_head<<<1, 256, 0, stream>>>(H32, Wd1, bd1, Wd2, bd2, (float*)d_out);

  // GAT branch -> d_out[35..128035)
  k_gat<<<(SEQ * 5) / GPB, 128, 0, stream>>>(x, adj, Wg, ag, Wgo, ago,
                                             (float*)d_out + BATCH * 7);
}
